// HMHSAEnergy_4234837754299
// MI455X (gfx1250) — matrix operand order repaired and checked by reading, not compiled, not run on hardware
//
#include <hip/hip_runtime.h>

// ---------------- problem constants ----------------
#define NN 3072
#define FF 512
#define EE 512
#define HH 8
#define DD 64
#define SCALE 0.125f      // HEAD_DIM^-0.5, D=64
#define NEGV  -1.0e30f

typedef __attribute__((ext_vector_type(16))) _Float16 v16h;
typedef __attribute__((ext_vector_type(8)))  _Float16 v8h;
typedef __attribute__((ext_vector_type(8)))  float    v8f;
typedef __attribute__((ext_vector_type(4)))  unsigned int u32x4;
typedef __attribute__((ext_vector_type(8)))  int i32x8;
typedef __attribute__((ext_vector_type(4)))  int i32x4;

#if defined(__has_builtin)
#if __has_builtin(__builtin_amdgcn_tensor_load_to_lds) && \
    __has_builtin(__builtin_amdgcn_s_wait_tensorcnt)
#define HAVE_TDM 1
#endif
#endif
#ifndef HAVE_TDM
#define HAVE_TDM 0
#endif

union AFrag { v16h v; v8h h[2]; };

__device__ __forceinline__ v8f vzero() {
    v8f z = {0.f, 0.f, 0.f, 0.f, 0.f, 0.f, 0.f, 0.f};
    return z;
}

__device__ __forceinline__ v8f wmma_f16(const AFrag& a, const AFrag& b, v8f c) {
    // D = A(16x32 f16) * B(32x16 f16) + C(16x16 f32)
    return __builtin_amdgcn_wmma_f32_16x16x32_f16(
        false, a.v, false, b.v, (short)0, c, false, false);
}

#if HAVE_TDM
// Issue one 2-D f16 tile DMA: global (row-major, row_stride elems) -> LDS (packed).
// D# layout per CDNA5 ISA ch.8 (group0: count/lds/global/type; group1: dims/strides).
__device__ __forceinline__ void tdm_load_2d_f16(
        unsigned lds_byte_off, const _Float16* gptr,
        unsigned tile_x, unsigned tile_y,
        unsigned tensor_x, unsigned tensor_y, unsigned row_stride_elems) {
    unsigned long long ga = (unsigned long long)(size_t)(const void*)gptr;
    u32x4 g0;
    g0[0] = 1u;                                   // count=1 (valid user descriptor)
    g0[1] = lds_byte_off;                         // lds_addr (bytes)
    g0[2] = (unsigned)(ga & 0xffffffffu);         // global_addr[31:0]
    g0[3] = (unsigned)((ga >> 32) & 0x01ffffffu)  // global_addr[56:32]
          | (2u << 30);                           // type=2 ("image")
    i32x8 g1;
    g1[0] = (int)(1u << 16);                      // wg_mask=0, data_size=1 (2 bytes)
    g1[1] = (int)((tensor_x & 0xffffu) << 16);                    // tensor_dim0 lo
    g1[2] = (int)((tensor_x >> 16) & 0xffffu)
          | (int)((tensor_y & 0xffffu) << 16);                    // dim0 hi | dim1 lo
    g1[3] = (int)((tensor_y >> 16) & 0xffffu)
          | (int)((tile_x & 0xffffu) << 16);                      // dim1 hi | tile_dim0
    g1[4] = (int)(tile_y & 0xffffu);                              // tile_dim1, tile_dim2=0
    g1[5] = (int)row_stride_elems;                                // tensor_dim0_stride lo
    g1[6] = 0;                                                    // stride hi, dim1_stride=0
    g1[7] = 0;
    i32x4 z4 = {0, 0, 0, 0};
#if __clang_major__ >= 23
    i32x8 z8 = {0, 0, 0, 0, 0, 0, 0, 0};
    __builtin_amdgcn_tensor_load_to_lds(g0, g1, z4, z4, z8, 0);
#else
    __builtin_amdgcn_tensor_load_to_lds(g0, g1, z4, z4, 0);
#endif
}
__device__ __forceinline__ unsigned lds_off(const void* p) {
    // LDS aperture: low 32 bits of flat shared address are the LDS byte offset.
    return (unsigned)(size_t)p;
}
#endif

// ---------------- kernel 0: fp32 -> f16 conversion ----------------
__global__ void k_convert(const float* __restrict__ x,
                          const float* __restrict__ wq,
                          const float* __restrict__ wk,
                          _Float16* __restrict__ xh,
                          _Float16* __restrict__ wqh,
                          _Float16* __restrict__ wkh) {
    int i = blockIdx.x * blockDim.x + threadIdx.x;
    const int XT = NN * FF, WT = EE * FF;
    if (i < XT)                xh[i]            = (_Float16)x[i];
    else if (i < XT + WT)      wqh[i - XT]      = (_Float16)wq[i - XT];
    else if (i < XT + 2 * WT)  wkh[i - XT - WT] = (_Float16)wk[i - XT - WT];
}

// ---------------- kernel 1: Q = x*WQ^T, K = x*WK^T (WMMA) ----------------
__global__ void __launch_bounds__(256)
k_proj(const _Float16* __restrict__ xh,
       const _Float16* __restrict__ wqh,
       const _Float16* __restrict__ wkh,
       _Float16* __restrict__ qh,
       _Float16* __restrict__ kh,
       _Float16* __restrict__ kth) {
    const int wave = threadIdx.x >> 5, lane = threadIdx.x & 31;
    const int half = lane >> 4, lm = lane & 15;
    const int m0 = blockIdx.x * 128 + wave * 16;
    const int n0 = blockIdx.y * 64;
    const bool isK = (blockIdx.z != 0);
    const _Float16* wh = isK ? wkh : wqh;
    _Float16* oh = isK ? kh : qh;

    v8f c[4];
    for (int i = 0; i < 4; ++i) c[i] = vzero();

    const int arow = m0 + lm;
    for (int k0 = 0; k0 < FF; k0 += 32) {
        AFrag a;
        const _Float16* ap = xh + arow * FF + k0 + half * 8;
        a.h[0] = *(const v8h*)ap;
        a.h[1] = *(const v8h*)(ap + 16);
        for (int nc = 0; nc < 4; ++nc) {
            AFrag b;
            const _Float16* bp = wh + (n0 + nc * 16 + lm) * FF + k0 + half * 8;
            b.h[0] = *(const v8h*)bp;
            b.h[1] = *(const v8h*)(bp + 16);
            c[nc] = wmma_f16(a, b, c[nc]);
        }
    }
    for (int nc = 0; nc < 4; ++nc) {
        int col = n0 + nc * 16 + lm;
        for (int r = 0; r < 8; ++r) {
            int row = m0 + half * 8 + r;
            _Float16 v = (_Float16)c[nc][r];
            oh[row * EE + col] = v;
            if (isK) kth[col * NN + row] = v;   // K^T: [E][N]
        }
    }
}

// ---------------- kernel 2: fused flash attention per head ----------------
// grid (H, N/128); 8 waves, wave owns 16 query rows. K/K^T tiles staged to
// LDS by the Tensor Data Mover, double-buffered on TENSORcnt.
__global__ void __launch_bounds__(256)
k_attn(const _Float16* __restrict__ qh,
       const _Float16* __restrict__ kh,
       const _Float16* __restrict__ kth,
       const unsigned char* __restrict__ adj,
       float* __restrict__ out,
       float* __restrict__ energies,
       float* __restrict__ lse_ws) {
    __shared__ _Float16 ldsP[8][16][32];      // per-wave P staging (f16)
    __shared__ float    ldsQN[8][16];         // per-wave 0.5*||q||^2
    __shared__ _Float16 ldsK[2][32][64];      // K tile  (keys x d), double buffered
    __shared__ _Float16 ldsKT[2][64][32];     // K^T tile (d x keys), double buffered

    const int wave = threadIdx.x >> 5, lane = threadIdx.x & 31;
    const int half = lane >> 4, lm = lane & 15;
    const int h  = blockIdx.x;
    const int q0 = blockIdx.y * 128 + wave * 16;

    AFrag aQ0, aQ1;
    {
        const _Float16* qp = qh + (q0 + lm) * EE + h * DD + half * 8;
        aQ0.h[0] = *(const v8h*)qp;        aQ0.h[1] = *(const v8h*)(qp + 16);
        aQ1.h[0] = *(const v8h*)(qp + 32); aQ1.h[1] = *(const v8h*)(qp + 48);
    }
    {
        float s = 0.f;
        for (int i = 0; i < 8; ++i) {
            float v0 = (float)aQ0.h[0][i], v1 = (float)aQ0.h[1][i];
            float v2 = (float)aQ1.h[0][i], v3 = (float)aQ1.h[1][i];
            s += v0 * v0 + v1 * v1 + v2 * v2 + v3 * v3;
        }
        s += __shfl_xor(s, 16, 32);
        if (half == 0) ldsQN[wave][lm] = 0.5f * s;
    }

    float m[8], l[8];
    for (int r = 0; r < 8; ++r) { m[r] = -3.0e38f; l[r] = 0.f; }
    v8f o[4];
    for (int i = 0; i < 4; ++i) o[i] = vzero();

#if HAVE_TDM
    if (wave == 0) {   // preload tile 0 into buffer 0 (2 descriptors)
        tdm_load_2d_f16(lds_off(&ldsK[0][0][0]),  kh + 0 * EE + h * DD,
                        64, 32, FF, NN, EE);
        tdm_load_2d_f16(lds_off(&ldsKT[0][0][0]), kth + (size_t)(h * DD) * NN + 0,
                        32, 64, NN, EE, NN);
    }
#endif

    for (int it = 0; it < NN / 32; ++it) {
        const int kt = it * 32;
        const int buf = it & 1;
#if HAVE_TDM
        if (wave == 0) {
            if (kt + 32 < NN) {
                tdm_load_2d_f16(lds_off(&ldsK[buf ^ 1][0][0]),
                                kh + (kt + 32) * EE + h * DD, 64, 32, FF, NN, EE);
                tdm_load_2d_f16(lds_off(&ldsKT[buf ^ 1][0][0]),
                                kth + (size_t)(h * DD) * NN + kt + 32, 32, 64, NN, EE, NN);
                __builtin_amdgcn_s_wait_tensorcnt((short)2);  // current tile resident
            } else {
                __builtin_amdgcn_s_wait_tensorcnt((short)0);
            }
        }
        __syncthreads();
#else
        __syncthreads();
        for (int t = threadIdx.x; t < 256; t += 256) {
            int n = t >> 3, d = (t & 7) * 8;
            *(v8h*)&ldsK[buf][n][d] = *(const v8h*)(kh + (kt + n) * EE + h * DD + d);
            int dd2 = t >> 2, n2 = (t & 3) * 8;
            *(v8h*)&ldsKT[buf][dd2][n2] =
                *(const v8h*)(kth + (size_t)(h * DD + dd2) * NN + kt + n2);
        }
        __syncthreads();
#endif
        // scores S = Q * Ktile^T (B-fragments from LDS K tile)
        v8f s[2];
        for (int c = 0; c < 2; ++c) {
            AFrag b0, b1;
            const _Float16* kp = &ldsK[buf][c * 16 + lm][half * 8];
            b0.h[0] = *(const v8h*)kp;        b0.h[1] = *(const v8h*)(kp + 16);
            b1.h[0] = *(const v8h*)(kp + 32); b1.h[1] = *(const v8h*)(kp + 48);
            v8f t = vzero();
            t = wmma_f16(aQ0, b0, t);
            t = wmma_f16(aQ1, b1, t);
            s[c] = t;
        }
        // scale + mask
        float sm[2][8], mk[2][8];
        for (int c = 0; c < 2; ++c)
            for (int r = 0; r < 8; ++r) {
                int q   = q0 + half * 8 + r;
                int key = kt + c * 16 + lm;
                float msk = adj[q * NN + key] ? 1.f : 0.f;
                mk[c][r] = msk;
                sm[c][r] = (msk != 0.f) ? s[c][r] * SCALE : NEGV;
            }
        // online softmax
        float newm[8], corr[8];
        for (int r = 0; r < 8; ++r) {
            float mx = fmaxf(sm[0][r], sm[1][r]);
            for (int i = 1; i < 16; i <<= 1) mx = fmaxf(mx, __shfl_xor(mx, i, 16));
            float nm = fmaxf(m[r], mx);
            corr[r] = __expf(m[r] - nm);
            newm[r] = nm;
        }
        float p[2][8];
        for (int c = 0; c < 2; ++c)
            for (int r = 0; r < 8; ++r)
                p[c][r] = mk[c][r] * __expf(sm[c][r] - newm[r]);
        for (int r = 0; r < 8; ++r) {
            float rs = p[0][r] + p[1][r];
            for (int i = 1; i < 16; i <<= 1) rs += __shfl_xor(rs, i, 16);
            l[r] = l[r] * corr[r] + rs;
            m[r] = newm[r];
        }
        for (int dc = 0; dc < 4; ++dc)
            for (int r = 0; r < 8; ++r)
                o[dc][r] *= corr[r];

        // P (C-layout f32) -> f16 A-fragment via per-wave LDS staging
        for (int c = 0; c < 2; ++c)
            for (int r = 0; r < 8; ++r)
                ldsP[wave][half * 8 + r][c * 16 + lm] = (_Float16)p[c][r];
        asm volatile("s_wait_dscnt 0" ::: "memory");
        AFrag aP;
        {
            const _Float16* pp = &ldsP[wave][lm][half * 8];
            aP.h[0] = *(const v8h*)pp;
            aP.h[1] = *(const v8h*)(pp + 16);
        }
        // O += P * Ktile (B-fragments from LDS K^T tile)
        for (int dc = 0; dc < 4; ++dc) {
            AFrag bV;
            const _Float16* vp = &ldsKT[buf][dc * 16 + lm][half * 8];
            bV.h[0] = *(const v8h*)vp;
            bV.h[1] = *(const v8h*)(vp + 16);
            o[dc] = wmma_f16(aP, bV, o[dc]);
        }
        __syncthreads();   // protect buf^1 before next iteration's DMA overwrites it
    }

    float invl[8];
    for (int r = 0; r < 8; ++r) invl[r] = 1.f / l[r];
    for (int dc = 0; dc < 4; ++dc)
        for (int r = 0; r < 8; ++r) {
            int row = q0 + half * 8 + r;
            out[row * EE + h * DD + dc * 16 + lm] = o[dc][r] * invl[r];
        }
    asm volatile("s_wait_dscnt 0" ::: "memory");
    for (int r = 0; r < 8; ++r) {
        int row = q0 + half * 8 + r;
        float lse = m[r] + __logf(l[r]);
        if (lm == 0) {
            lse_ws[h * NN + row]   = lse;
            energies[h * NN + row] = -lse * 8.0f + ldsQN[wave][half * 8 + r];
        }
    }
}

// ---------------- kernel 3: attn_mean (score recompute, sum over heads) ----
// grid (N/128, N/128); K slice per head staged via TDM, double buffered.
__global__ void __launch_bounds__(256)
k_attnmean(const _Float16* __restrict__ qh,
           const _Float16* __restrict__ kh,
           const float* __restrict__ lse_ws,
           const unsigned char* __restrict__ adj,
           float* __restrict__ attn_mean) {
    __shared__ _Float16 ldsK3[2][128][64];    // K slice (keys x d), 2 x 16KB

    const int wave = threadIdx.x >> 5, lane = threadIdx.x & 31;
    const int half = lane >> 4, lm = lane & 15;
    const int q0 = blockIdx.x * 128 + wave * 16;
    const int k0 = blockIdx.y * 128;

    unsigned long long mbits = 0ull;
    for (int kc = 0; kc < 8; ++kc)
        for (int r = 0; r < 8; ++r) {
            int q = q0 + half * 8 + r;
            int key = k0 + kc * 16 + lm;
            if (adj[q * NN + key]) mbits |= 1ull << (kc * 8 + r);
        }

    v8f acc[8];
    for (int i = 0; i < 8; ++i) acc[i] = vzero();

#if HAVE_TDM
    if (wave == 0)
        tdm_load_2d_f16(lds_off(&ldsK3[0][0][0]), kh + k0 * EE + 0 * DD,
                        64, 128, FF, NN, EE);
#endif

    for (int h = 0; h < HH; ++h) {
        const int buf = h & 1;
#if HAVE_TDM
        if (wave == 0) {
            if (h + 1 < HH) {
                tdm_load_2d_f16(lds_off(&ldsK3[buf ^ 1][0][0]),
                                kh + k0 * EE + (h + 1) * DD, 64, 128, FF, NN, EE);
                __builtin_amdgcn_s_wait_tensorcnt((short)1);
            } else {
                __builtin_amdgcn_s_wait_tensorcnt((short)0);
            }
        }
        __syncthreads();
#else
        __syncthreads();
        for (int t = threadIdx.x; t < 1024; t += 256) {
            int n = t >> 3, d = (t & 7) * 8;
            *(v8h*)&ldsK3[buf][n][d] = *(const v8h*)(kh + (k0 + n) * EE + h * DD + d);
        }
        __syncthreads();
#endif
        AFrag aQ0, aQ1;
        const _Float16* qp = qh + (q0 + lm) * EE + h * DD + half * 8;
        aQ0.h[0] = *(const v8h*)qp;        aQ0.h[1] = *(const v8h*)(qp + 16);
        aQ1.h[0] = *(const v8h*)(qp + 32); aQ1.h[1] = *(const v8h*)(qp + 48);
        float lse_r[8];
        for (int r = 0; r < 8; ++r)
            lse_r[r] = lse_ws[h * NN + q0 + half * 8 + r];
        for (int kc = 0; kc < 8; ++kc) {
            AFrag b0, b1;
            const _Float16* kp = &ldsK3[buf][kc * 16 + lm][half * 8];
            b0.h[0] = *(const v8h*)kp;        b0.h[1] = *(const v8h*)(kp + 16);
            b1.h[0] = *(const v8h*)(kp + 32); b1.h[1] = *(const v8h*)(kp + 48);
            v8f t = vzero();
            t = wmma_f16(aQ0, b0, t);
            t = wmma_f16(aQ1, b1, t);
            for (int r = 0; r < 8; ++r) {
                float pv = 0.f;
                if (mbits & (1ull << (kc * 8 + r)))
                    pv = __expf(t[r] * SCALE - lse_r[r]);
                acc[kc][r] += pv;
            }
        }
        __syncthreads();   // all waves done with buf before it is re-filled
    }
    for (int kc = 0; kc < 8; ++kc)
        for (int r = 0; r < 8; ++r) {
            int q = q0 + half * 8 + r;
            int key = k0 + kc * 16 + lm;
            attn_mean[(size_t)q * NN + key] = acc[kc][r] * 0.125f;
        }
}

// ---------------- launch ----------------
extern "C" void kernel_launch(void* const* d_in, const int* in_sizes, int n_in,
                              void* d_out, int out_size, void* d_ws, size_t ws_size,
                              hipStream_t stream) {
    const float* x           = (const float*)d_in[0];
    const unsigned char* adj = (const unsigned char*)d_in[1];  // jnp bool -> 1 byte
    const float* wq          = (const float*)d_in[2];
    const float* wk          = (const float*)d_in[3];

    constexpr size_t SZ_XH = (size_t)NN * FF * 2;
    constexpr size_t SZ_WH = (size_t)EE * FF * 2;
    constexpr size_t SZ_QH = (size_t)NN * EE * 2;
    char* w = (char*)d_ws;
    _Float16* xh  = (_Float16*)(w);
    _Float16* wqh = (_Float16*)(w + SZ_XH);
    _Float16* wkh = (_Float16*)(w + SZ_XH + SZ_WH);
    _Float16* qhp = (_Float16*)(w + SZ_XH + 2 * SZ_WH);
    _Float16* khp = (_Float16*)(w + SZ_XH + 2 * SZ_WH + SZ_QH);
    _Float16* kth = (_Float16*)(w + SZ_XH + 2 * SZ_WH + 2 * SZ_QH);
    float*    lse = (float*)  (w + SZ_XH + 2 * SZ_WH + 3 * SZ_QH);

    float* out       = (float*)d_out;                 // [N, E]
    float* attn_mean = out + (size_t)NN * EE;         // [N, N]
    float* energies  = attn_mean + (size_t)NN * NN;   // [H, N]

    k_convert<<<(NN * FF + 2 * EE * FF + 255) / 256, 256, 0, stream>>>(
        x, wq, wk, xh, wqh, wkh);
    k_proj<<<dim3(NN / 128, EE / 64, 2), 256, 0, stream>>>(
        xh, wqh, wkh, qhp, khp, kth);
    k_attn<<<dim3(HH, NN / 128), 256, 0, stream>>>(
        qhp, khp, kth, adj, out, energies, lse);
    k_attnmean<<<dim3(NN / 128, NN / 128), 256, 0, stream>>>(
        qhp, khp, lse, adj, attn_mean);
}
